// RadiomicsExpert_17291538334409
// MI455X (gfx1250) — matrix operand order repaired and checked by reading, not compiled, not run on hardware
//
#include <hip/hip_runtime.h>
#include <hip/hip_bf16.h>
#include <math.h>

// Problem dims (from reference)
#define BSZ    8
#define DEPTH  64
#define HH     256
#define WW     256
#define NPIX   (HH * WW)          // 65536 per slice
#define NSLICE (BSZ * DEPTH)      // 512
#define EMB    256
#define NFEAT  18
#define NBINS  16

typedef __attribute__((ext_vector_type(16))) _Float16 v16h;
typedef __attribute__((ext_vector_type(8)))  float    v8f;

// ---------- float <-> order-preserving uint ----------
__device__ __forceinline__ unsigned f2ord(float f) {
  unsigned u = __float_as_uint(f);
  return u ^ (((unsigned)((int)u >> 31)) | 0x80000000u);
}
__device__ __forceinline__ float ord2f(unsigned u) {
  unsigned m = (u & 0x80000000u) ? 0x80000000u : 0xFFFFFFFFu;
  return __uint_as_float(u ^ m);
}

// ---------- block reduction (double) ----------
__device__ double blockReduceSum(double v) {
  __shared__ double tmp[32];
  const int lane = threadIdx.x & 31;
  const int wid  = threadIdx.x >> 5;
  #pragma unroll
  for (int o = 16; o > 0; o >>= 1) v += __shfl_down(v, o, 32);
  if (lane == 0) tmp[wid] = v;
  __syncthreads();
  const int nw = (blockDim.x + 31) >> 5;
  v = (threadIdx.x < (unsigned)nw) ? tmp[threadIdx.x] : 0.0;
  if (wid == 0) {
    #pragma unroll
    for (int o = 16; o > 0; o >>= 1) v += __shfl_down(v, o, 32);
  }
  if (threadIdx.x == 0) tmp[0] = v;
  __syncthreads();
  double r = tmp[0];
  __syncthreads();
  return r;
}

// ---------- exact k-th smallest of NPIX floats living in LDS ----------
// 4 passes of 8-bit MSD->LSD radix narrowing with a 256-bin LDS histogram.
// All threads of the block must call (contains __syncthreads).
__device__ float radix_select(const float* sdata, unsigned* hist, int k) {
  const int tid = threadIdx.x, nt = blockDim.x;
  unsigned prefix = 0;
  int rank = k;
  for (int shift = 24; shift >= 0; shift -= 8) {
    for (int i = tid; i < 256; i += nt) hist[i] = 0;
    __syncthreads();
    const unsigned himask = (shift == 24) ? 0u : (0xFFFFFFFFu << (shift + 8));
    for (int i = tid; i < NPIX; i += nt) {
      unsigned u = f2ord(sdata[i]);
      if ((u & himask) == prefix) atomicAdd(&hist[(u >> shift) & 0xFFu], 1u);
    }
    __syncthreads();
    // every thread computes the same scan result (broadcast LDS reads)
    int c = 0, b = 0;
    for (; b < 256; ++b) {
      int hv = (int)hist[b];
      if (rank < c + hv) break;
      c += hv;
    }
    prefix |= ((unsigned)b) << shift;
    rank -= c;
    __syncthreads();
  }
  return ord2f(prefix);
}

// =====================================================================
// Kernel 1: one workgroup per slice; slice staged into 256 KB of LDS.
// =====================================================================
extern "C" __global__ __launch_bounds__(512)
void k_features(const float* __restrict__ vol, float* __restrict__ feats) {
  extern __shared__ float sdata[];          // NPIX floats (256 KB dynamic LDS)
  __shared__ unsigned hist[256];
  __shared__ unsigned u_max, u_cnt;
  __shared__ int hbin[NBINS];

  const int tid = threadIdx.x, nt = blockDim.x;
  const int slice = blockIdx.x;
  const float* __restrict__ src = vol + (size_t)slice * NPIX;

  if (tid == 0) { u_max = 0u; u_cnt = 0u; }
  __syncthreads();

  // ---- stage slice to LDS; fused ordered-max reduction ----
  unsigned mloc = 0u;
  for (int i = tid; i < NPIX; i += nt) {
    __builtin_prefetch(src + i + 8 * nt, 0, 1);   // global_prefetch_b8
    float v = src[i];
    sdata[i] = v;
    unsigned o = f2ord(v);
    mloc = (o > mloc) ? o : mloc;
  }
  atomicMax(&u_max, mloc);
  __syncthreads();

  // ---- 80th percentile threshold (linear-interp quantile, f32 math) ----
  const float posT = 0.8f * (float)(NPIX - 1);
  const int   loT  = (int)floorf(posT);
  const float frT  = posT - floorf(posT);
  float thresh = radix_select(sdata, hist, loT);
  if (frT > 0.0f) {                 // uniform branch
    int hiT = loT + 1; if (hiT > NPIX - 1) hiT = NPIX - 1;
    float vhiT = radix_select(sdata, hist, hiT);
    thresh = thresh * (1.0f - frT) + vhiT * frT;
  }

  // ---- masked count ----
  {
    int cl = 0;
    for (int i = tid; i < NPIX; i += nt) cl += (sdata[i] >= thresh) ? 1 : 0;
    atomicAdd(&u_cnt, (unsigned)cl);
  }
  __syncthreads();
  const int   n    = (int)u_cnt;
  const int   base = NPIX - n;       // masked-sorted == tail of full-sorted
  const float fn   = (float)n;
  const float vmax = ord2f(u_max);
  const float vmin = thresh;         // smallest value passing `>= thresh`

  // ---- masked quantiles via rank translation ----
  float q[3];
  const float ps[3] = {0.25f, 0.5f, 0.75f};
  for (int qi = 0; qi < 3; ++qi) {
    float pos = ps[qi] * (fn - 1.0f);
    float lo  = floorf(pos);
    float fr  = pos - lo;
    int lo_i = (int)lo;
    int hi_i = lo_i + 1; if (hi_i > n - 1) hi_i = n - 1;   // uniform in block
    float vlo = radix_select(sdata, hist, base + lo_i);
    float vhi = (hi_i == lo_i) ? vlo : radix_select(sdata, hist, base + hi_i);
    q[qi] = vlo * (1.0f - fr) + vhi * fr;
  }

  // ---- masked raw moments + abs + centroid (double accumulation) ----
  double s1 = 0, s2 = 0, s3 = 0, s4 = 0, sa = 0, sy = 0, sx = 0;
  for (int i = tid; i < NPIX; i += nt) {
    float v = sdata[i];
    if (v >= thresh) {
      double d = (double)v, d2 = d * d;
      s1 += d; s2 += d2; s3 += d2 * d; s4 += d2 * d2;
      sa += fabs(d);
      sy += (double)(i >> 8);
      sx += (double)(i & 255);
    }
  }
  s1 = blockReduceSum(s1); s2 = blockReduceSum(s2);
  s3 = blockReduceSum(s3); s4 = blockReduceSum(s4);
  sa = blockReduceSum(sa); sy = blockReduceSum(sy); sx = blockReduceSum(sx);

  // ---- gradient-magnitude stats (np.gradient semantics) ----
  double sg = 0, sg2 = 0;
  for (int i = tid; i < NPIX; i += nt) {
    float v = sdata[i];
    if (v >= thresh) {
      int y = i >> 8, x = i & 255;
      float gy = (y == 0)      ? sdata[i + WW] - v
               : (y == HH - 1) ? v - sdata[i - WW]
                               : 0.5f * (sdata[i + WW] - sdata[i - WW]);
      float gx = (x == 0)      ? sdata[i + 1] - v
               : (x == WW - 1) ? v - sdata[i - 1]
                               : 0.5f * (sdata[i + 1] - sdata[i - 1]);
      float gm = sqrtf(gy * gy + gx * gx);
      sg += (double)gm; sg2 += (double)gm * (double)gm;
    }
  }
  sg = blockReduceSum(sg); sg2 = blockReduceSum(sg2);

  // ---- 16-bin masked histogram (entropy) ----
  if (tid < NBINS) hbin[tid] = 0;
  __syncthreads();
  {
    float rng   = vmax - vmin;
    float denom = (rng > 0.0f) ? rng : 1.0f;
    for (int i = tid; i < NPIX; i += nt) {
      float v = sdata[i];
      if (v >= thresh) {
        int bin = (int)floorf((v - vmin) / denom * (float)NBINS);
        bin = bin < 0 ? 0 : (bin > NBINS - 1 ? NBINS - 1 : bin);
        atomicAdd(&hbin[bin], 1);
      }
    }
  }
  __syncthreads();

  if (tid == 0) {
    const double dn = (double)n;
    const double mu = s1 / dn;
    const double r2 = s2 / dn, r3 = s3 / dn, r4 = s4 / dn;
    const double m2 = r2 - mu * mu;
    const double m3 = r3 - 3.0 * mu * r2 + 2.0 * mu * mu * mu;
    const double m4 = r4 - 4.0 * mu * r3 + 6.0 * mu * mu * r2 - 3.0 * mu * mu * mu * mu;

    float mean  = (float)mu;
    float stdv  = sqrtf(fmaxf((float)m2, 0.0f));
    float se    = fmaxf(stdv, 1e-6f);
    float skew  = fminf(fmaxf((float)m3 / (se * se * se), -50.0f), 50.0f);
    float kurt  = fminf(fmaxf((float)m4 / (se * se * se * se), 0.0f), 100.0f);
    float msq   = (float)r2;
    float amean = (float)(sa / dn);

    float hsum = 0.0f;
    for (int b2 = 0; b2 < NBINS; ++b2) hsum += (float)hbin[b2];
    float hden = fmaxf(hsum, 1.0f);
    float ent = 0.0f;
    for (int b2 = 0; b2 < NBINS; ++b2) {
      float p = fmaxf((float)hbin[b2] / hden, 1e-6f);
      ent -= p * logf(p);
    }
    if (fabsf(vmin - vmax) <= (1e-8f + 1e-5f * fabsf(vmax))) ent = 0.0f;

    float gmean = (float)(sg / dn);
    double gv   = sg2 / dn - (sg / dn) * (sg / dn);
    float gstd  = sqrtf(fmaxf((float)gv, 0.0f));
    float cy    = (float)(sy / dn) / (float)(HH - 1);
    float cx    = (float)(sx / dn) / (float)(WW - 1);
    float frac  = fn / (float)NPIX;

    float* o = feats + slice * NFEAT;
    o[0]  = mean;  o[1]  = stdv;  o[2]  = vmin;  o[3]  = vmax;
    o[4]  = q[0];  o[5]  = q[1];  o[6]  = q[2];  o[7]  = msq;
    o[8]  = ent;   o[9]  = skew;  o[10] = kurt;  o[11] = frac;
    o[12] = gmean; o[13] = gstd;  o[14] = cy;    o[15] = cx;
    o[16] = frac;  o[17] = amean;
  }
}

// =====================================================================
// Kernel 2: feats(512x18) @ W(18x256) + b  ->  LayerNorm(gamma,beta)
// WMMA f32_16x16x32_f16, K padded 18->32 with zeros.
// Block = 256 threads (8 waves), each block covers a 16-row stripe.
// =====================================================================
extern "C" __global__ __launch_bounds__(256)
void k_gemm_ln(const float* __restrict__ feats, const float* __restrict__ Wm,
               const float* __restrict__ bias,  const float* __restrict__ gamma,
               const float* __restrict__ beta,  float* __restrict__ out,
               int out_size) {
  __shared__ float xs[16][EMB];      // 16 KB
  __shared__ float mu_s[16], rs_s[16];

  const int tid  = threadIdx.x;
  const int lane = tid & 31;
  const int wv   = tid >> 5;         // 0..7
  const int R0   = blockIdx.x * 16;

  const int half = (lane >> 4) & 1;  // lane group
  const int mn   = lane & 15;        // M (for A/D rows) or N (for B/D cols)

  // ---- A fragment: 16x32 f16, rows R0..R0+15, K padded to 32 ----
  // half 0: elems 0..7 -> K0..7,  elems 8..15 -> K16..23
  // half 1: elems 0..7 -> K8..15, elems 8..15 -> K24..31
  v16h afrag;
  #pragma unroll
  for (int e = 0; e < 16; ++e) {
    int K = e + ((e < 8) ? (half ? 8 : 0) : (half ? 16 : 8));
    float v = (K < NFEAT) ? feats[(R0 + mn) * NFEAT + K] : 0.0f;
    afrag[e] = (_Float16)v;
  }

  // ---- each wave computes two 16-col tiles ----
  #pragma unroll
  for (int jj = 0; jj < 2; ++jj) {
    const int C0 = (wv * 2 + jj) * 16;
    // B fragment: 32x16 f16; same K layout as A (half 0: K0..7,K16..23; half 1: K8..15,K24..31)
    v16h bfrag;
    #pragma unroll
    for (int e = 0; e < 16; ++e) {
      int K = e + ((e < 8) ? (half ? 8 : 0) : (half ? 16 : 8));
      float v = (K < NFEAT) ? Wm[K * EMB + C0 + mn] : 0.0f;
      bfrag[e] = (_Float16)v;
    }
    v8f c = {0.f, 0.f, 0.f, 0.f, 0.f, 0.f, 0.f, 0.f};
    c = __builtin_amdgcn_wmma_f32_16x16x32_f16(
            /*neg_a=*/false, afrag, /*neg_b=*/false, bfrag,
            /*c_mod=*/(short)0, c, /*reuse_a=*/false, /*reuse_b=*/false);
    // D: VGPR r -> M = r (half 0) / 8+r (half 1), N = mn
    #pragma unroll
    for (int r = 0; r < 8; ++r) {
      int m   = (half ? 8 : 0) + r;
      int col = C0 + mn;
      xs[m][col] = c[r] + bias[col];
    }
  }
  __syncthreads();

  // ---- LayerNorm over EMB per row ----
  if (tid < 16) {
    float s = 0.0f;
    for (int c2 = 0; c2 < EMB; ++c2) s += xs[tid][c2];
    float mu = s / (float)EMB;
    float v = 0.0f;
    for (int c2 = 0; c2 < EMB; ++c2) { float d = xs[tid][c2] - mu; v += d * d; }
    v /= (float)EMB;
    mu_s[tid] = mu;
    rs_s[tid] = rsqrtf(v + 1e-5f);
  }
  __syncthreads();

  for (int idx = tid; idx < 16 * EMB; idx += blockDim.x) {
    int r = idx >> 8, c2 = idx & 255;
    float val = (xs[r][c2] - mu_s[r]) * rs_s[r] * gamma[c2] + beta[c2];
    out[(size_t)(R0 + r) * EMB + c2] = val;
  }

  // ---- padding mask (all false -> 0.0), if the harness allotted space ----
  if (out_size >= NSLICE * EMB + NSLICE && tid < 16)
    out[NSLICE * EMB + R0 + tid] = 0.0f;
}

// =====================================================================
extern "C" void kernel_launch(void* const* d_in, const int* in_sizes, int n_in,
                              void* d_out, int out_size, void* d_ws, size_t ws_size,
                              hipStream_t stream) {
  (void)in_sizes; (void)n_in; (void)ws_size;
  const float* vol   = (const float*)d_in[0];  // (8,64,256,256)
  const float* Wm    = (const float*)d_in[1];  // (18,256)
  const float* bias  = (const float*)d_in[2];  // (256,)
  const float* gamma = (const float*)d_in[3];  // (256,)
  const float* beta  = (const float*)d_in[4];  // (256,)
  float* feats = (float*)d_ws;                 // 512*18 f32 scratch

  const size_t shmem = (size_t)NPIX * sizeof(float);   // 256 KB of 320 KB LDS
  k_features<<<NSLICE, 512, shmem, stream>>>(vol, feats);
  k_gemm_ln<<<NSLICE / 16, 256, 0, stream>>>(feats, Wm, bias, gamma, beta,
                                             (float*)d_out, out_size);
}
